// DNNF_26259430048006
// MI455X (gfx1250) — compile-verified, matrix operand order checked by reading, not run on hardware
//
#include <hip/hip_runtime.h>

// Problem constants (mirror the reference)
#define BATCH_N   8192
#define DIM       512
#define NLIT      10752
#define NCONJ     2688
#define NFORM     256
#define LTILES    (NLIT / 16)   // 672 column tiles for the literal GEMM
#define FTILES    (NFORM / 16)  // 16 column tiles for the mu GEMM
#define KBLK      (DIM / 32)    // 16 K-steps of 32 per tile
#define ROWS      16            // M-tile rows per block
#define MAIN_T    256           // 8 wave32 per block

typedef __attribute__((ext_vector_type(16))) __bf16 v16bf;
typedef __attribute__((ext_vector_type(8)))  float  v8f;

union Frag16 { uint4 q[2]; v16bf v; };   // 32 bytes = one lane's 16 bf16 fragment

__device__ __forceinline__ unsigned short f2bf(float f) {
  unsigned u = __float_as_uint(f);
  u += 0x7FFFu + ((u >> 16) & 1u);       // round-to-nearest-even
  return (unsigned short)(u >> 16);
}

// Position of element (kr, c) of a 32(K) x 16(N) bf16 fragment, flattened as
// [lane][16]: lanes 0-15 hold K {0..7, 16..23}, lanes 16-31 hold K {8..15, 24..31}.
__device__ __forceinline__ int frag_pos(int kr, int c) {
  int h, idx;
  if (kr < 16) { h = kr >> 3;               idx = kr & 7; }
  else         { int k2 = kr - 16; h = k2 >> 3; idx = 8 + (k2 & 7); }
  return (c + 16 * h) * 16 + idx;
}

// ---------------- pack kernels (fp32 -> bf16, WMMA-fragment swizzle) ----------------

// weight/mask are [DIM, NLIT] row-major; Wp is [LTILES][KBLK][32 lanes][16] bf16.
__global__ void pack_weff(const float* __restrict__ w, const float* __restrict__ m,
                          unsigned short* __restrict__ wp) {
  int tid = blockIdx.x * blockDim.x + threadIdx.x;
  if (tid >= DIM * NLIT) return;
  int k = tid / NLIT, l = tid - k * NLIT;
  float v = w[tid] * m[tid];
  int lt = l >> 4, c = l & 15;
  int kb = k >> 5, kr = k & 31;
  wp[((lt * KBLK + kb) * 32) * 16 + frag_pos(kr, c)] = f2bf(v);
}

// mu is [NFORM, DIM] row-major; Mup same fragment layout (B-operand = mu^T).
__global__ void pack_mu(const float* __restrict__ mu, unsigned short* __restrict__ mup) {
  int tid = blockIdx.x * blockDim.x + threadIdx.x;
  if (tid >= NFORM * DIM) return;
  int f = tid >> 9, k = tid & (DIM - 1);
  int lt = f >> 4, c = f & 15;
  int kb = k >> 5, kr = k & 31;
  mup[((lt * KBLK + kb) * 32) * 16 + frag_pos(kr, c)] = f2bf(mu[tid]);
}

__global__ void mu_sq(const float* __restrict__ mu, float* __restrict__ musq) {
  int f = threadIdx.x;
  float s = 0.f;
  const float* r = mu + f * DIM;
  for (int k = 0; k < DIM; ++k) { float v = r[k]; s += v * v; }
  musq[f] = s;
}

// ---------------- fused main kernel ----------------

__global__ __launch_bounds__(MAIN_T)
void dnnf_fused(const float* __restrict__ x,
                const unsigned short* __restrict__ wp,
                const unsigned short* __restrict__ mup,
                const float* __restrict__ bias,
                const float* __restrict__ sigma,
                const float* __restrict__ musq,
                const int* __restrict__ lit2conj,
                const int* __restrict__ conj2form,
                float* __restrict__ out) {
  extern __shared__ char smem[];
  float*          conjAcc = (float*)smem;                             // [ROWS][NCONJ]
  float*          formAcc = conjAcc + ROWS * NCONJ;                   // [ROWS][NFORM]
  float*          locAcc  = formAcc + ROWS * NFORM;                   // [ROWS][NFORM]
  unsigned short* xt      = (unsigned short*)(locAcc + ROWS * NFORM); // [ROWS][DIM] bf16
  float*          xsq     = (float*)(xt + ROWS * DIM);                // [ROWS]
  float*          rowsum  = xsq + ROWS;                               // [ROWS]

  const int tid  = threadIdx.x;
  const int row0 = blockIdx.x * ROWS;

  for (int i = tid; i < ROWS * NCONJ; i += MAIN_T) conjAcc[i] = 0.f;
  for (int i = tid; i < ROWS * NFORM; i += MAIN_T) formAcc[i] = 0.f;
  if (tid < ROWS) { xsq[tid] = 0.f; rowsum[tid] = 0.f; }
  __syncthreads();

  // Stage X tile (fp32 -> bf16) and row |x|^2 partials.
  {
    int row = tid >> 4, chunk = tid & 15;            // 16 threads per row, 32 elems each
    const float* xr = x + (size_t)(row0 + row) * DIM + chunk * 32;
    int base = row * DIM + chunk * 32;
    float s = 0.f;
#pragma unroll
    for (int j = 0; j < 32; ++j) { float v = xr[j]; s += v * v; xt[base + j] = f2bf(v); }
    atomicAdd(&xsq[row], s);
  }
  __syncthreads();

  const int wave  = tid >> 5, lane = tid & 31;
  const int col16 = lane & 15, hh = lane >> 4;
  const uint4* wq = (const uint4*)wp;
  const uint4* mq = (const uint4*)mup;

  // Hoist ALL A fragments (full K=512 of this wave's 16 rows) into registers:
  // 16 kb-steps x 32B/lane = 128 VGPRs, loaded once instead of per column tile.
  Frag16 afr[KBLK];
#pragma unroll
  for (int kb = 0; kb < KBLK; ++kb) {
    const uint4* aq = (const uint4*)(xt + (col16 * DIM + kb * 32 + hh * 8));
    afr[kb].q[0] = aq[0];
    afr[kb].q[1] = aq[2];                              // +16 bf16 elements
  }

  // ---- literal GEMM: tanh(X @ Weff + b) scattered into conjunction sums ----
  // Two column tiles per iteration -> two independent WMMA accumulation chains.
  for (int lt = 2 * wave; lt < LTILES; lt += 16) {
    v8f acc0 = {0.f, 0.f, 0.f, 0.f, 0.f, 0.f, 0.f, 0.f};
    v8f acc1 = {0.f, 0.f, 0.f, 0.f, 0.f, 0.f, 0.f, 0.f};
    const int col0 = lt * 16 + col16, col1 = col0 + 16;
    const int cj0 = lit2conj[col0],   cj1 = lit2conj[col1];
    const float bv0 = bias[col0],     bv1 = bias[col1];
#pragma unroll
    for (int kb = 0; kb < KBLK; ++kb) {
      Frag16 b0, b1;
      int fb0 = ((lt * KBLK + kb) * 32 + lane) * 2;          // uint4 units
      int fb1 = (((lt + 1) * KBLK + kb) * 32 + lane) * 2;
      b0.q[0] = wq[fb0]; b0.q[1] = wq[fb0 + 1];
      b1.q[0] = wq[fb1]; b1.q[1] = wq[fb1 + 1];
      acc0 = __builtin_amdgcn_wmma_f32_16x16x32_bf16(false, afr[kb].v, false, b0.v,
                                                     (short)0, acc0, false, false);
      acc1 = __builtin_amdgcn_wmma_f32_16x16x32_bf16(false, afr[kb].v, false, b1.v,
                                                     (short)0, acc1, false, false);
    }
    // C/D layout: VGPR r -> row r + 8*hh, column col16.
#pragma unroll
    for (int r = 0; r < 8; ++r) {
      int row = r + 8 * hh;
      atomicAdd(&conjAcc[row * NCONJ + cj0], tanhf(acc0[r] + bv0));
      atomicAdd(&conjAcc[row * NCONJ + cj1], tanhf(acc1[r] + bv1));
    }
  }
  __syncthreads();

  // ---- conj -> formula: tanh(sum - depth + 1.5) scattered into formula sums ----
  for (int i = tid; i < ROWS * NCONJ; i += MAIN_T) {
    int row = i / NCONJ, c = i - row * NCONJ;
    float depth = (float)(2 * (c % 3 + 1));             // depths tile [2,4,6]
    float v = tanhf(conjAcc[i] - depth + 1.5f);
    atomicAdd(&formAcc[row * NFORM + conj2form[c]], v);
  }
  __syncthreads();

  // ---- localization GEMM: X @ mu^T, RBF, exp for softmax ----
  // 16 formula tiles over 8 waves: each wave does 2 tiles in one dual pass.
  {
    const int lt0 = 2 * wave, lt1 = 2 * wave + 1;
    v8f acc0 = {0.f, 0.f, 0.f, 0.f, 0.f, 0.f, 0.f, 0.f};
    v8f acc1 = {0.f, 0.f, 0.f, 0.f, 0.f, 0.f, 0.f, 0.f};
#pragma unroll
    for (int kb = 0; kb < KBLK; ++kb) {
      Frag16 b0, b1;
      int fb0 = ((lt0 * KBLK + kb) * 32 + lane) * 2;
      int fb1 = ((lt1 * KBLK + kb) * 32 + lane) * 2;
      b0.q[0] = mq[fb0]; b0.q[1] = mq[fb0 + 1];
      b1.q[0] = mq[fb1]; b1.q[1] = mq[fb1 + 1];
      acc0 = __builtin_amdgcn_wmma_f32_16x16x32_bf16(false, afr[kb].v, false, b0.v,
                                                     (short)0, acc0, false, false);
      acc1 = __builtin_amdgcn_wmma_f32_16x16x32_bf16(false, afr[kb].v, false, b1.v,
                                                     (short)0, acc1, false, false);
    }
    const int f0 = lt0 * 16 + col16, f1 = lt1 * 16 + col16;
    const float msq0 = musq[f0], msq1 = musq[f1];
    const float sg0 = sigma[f0], sg1 = sigma[f1];
    const float inv2s0 = 0.5f / (sg0 * sg0);
    const float inv2s1 = 0.5f / (sg1 * sg1);
#pragma unroll
    for (int r = 0; r < 8; ++r) {
      int row = r + 8 * hh;
      float xs = xsq[row];
      float sq0 = xs - 2.f * acc0[r] + msq0;
      float sq1 = xs - 2.f * acc1[r] + msq1;
      float e0 = expf(2.0f * expf(-inv2s0 * sq0));      // TEMPERATURE = 2; arg in (0,2]
      float e1 = expf(2.0f * expf(-inv2s1 * sq1));
      locAcc[row * NFORM + f0] = e0;
      locAcc[row * NFORM + f1] = e1;
      atomicAdd(&rowsum[row], e0 + e1);
    }
  }
  __syncthreads();

  // ---- epilogue: dnnf * softmax ----
  for (int i = tid; i < ROWS * NFORM; i += MAIN_T) {
    int row = i >> 8, f = i & (NFORM - 1);
    float orb  = (float)(6 + 3 * (f >> 6));             // conj-per-formula [6,9,12,15]
    float dnnf = tanhf(formAcc[i] + orb - 1.5f);
    out[(size_t)(row0 + row) * NFORM + f] = dnnf * (locAcc[i] / rowsum[row]);
  }
}

// ---------------- launcher ----------------

extern "C" void kernel_launch(void* const* d_in, const int* in_sizes, int n_in,
                              void* d_out, int out_size, void* d_ws, size_t ws_size,
                              hipStream_t stream) {
  const float* x      = (const float*)d_in[0];
  const float* weight = (const float*)d_in[1];
  const float* mask   = (const float*)d_in[2];
  const float* bias   = (const float*)d_in[3];
  const float* mu     = (const float*)d_in[4];
  const float* sigma  = (const float*)d_in[5];
  const int*   l2c    = (const int*)d_in[6];
  const int*   c2f    = (const int*)d_in[7];
  float*       out    = (float*)d_out;

  // Workspace carve-up (bf16 fragment-packed operands + |mu|^2)
  unsigned short* wp   = (unsigned short*)d_ws;                       // 10.5 MiB
  unsigned short* mup  = wp + (size_t)NLIT * DIM;                     // 256 KiB
  float*          musq = (float*)(mup + (size_t)NFORM * DIM);         // 1 KiB
  (void)ws_size; (void)n_in; (void)in_sizes; (void)out_size;

  pack_weff<<<(DIM * NLIT + 255) / 256, 256, 0, stream>>>(weight, mask, wp);
  pack_mu<<<(NFORM * DIM + 255) / 256, 256, 0, stream>>>(mu, mup);
  mu_sq<<<1, NFORM, 0, stream>>>(mu, musq);

  const int smemBytes = ROWS * NCONJ * 4      // conjAcc  172032
                      + ROWS * NFORM * 4 * 2  // formAcc + locAcc
                      + ROWS * DIM * 2        // bf16 X tile
                      + 2 * ROWS * 4;         // xsq + rowsum   => 221312 B
  hipFuncSetAttribute((const void*)dnnf_fused,
                      hipFuncAttributeMaxDynamicSharedMemorySize, smemBytes);
  dnnf_fused<<<BATCH_N / ROWS, MAIN_T, smemBytes, stream>>>(
      x, wp, mup, bias, sigma, musq, l2c, c2f, out);
}
